// LlamaMoEDecoderLayer_87943750352942
// MI455X (gfx1250) — matrix operand order repaired and checked by reading, not compiled, not run on hardware
//
#include <hip/hip_runtime.h>
#include <hip/hip_bf16.h>
#include <math.h>

// ---------------------------------------------------------------------------
// LlamaMoEDecoderLayer for MI455X (gfx1250, wave32, WMMA + TDM).
// - All GEMMs: bf16 WMMA, fp32 accum. Weights pre-transposed to [N,K] so both
//   A and B fragments are contiguous-K ds_load_b128 from LDS.
// - GEMM tiles staged by the Tensor Data Mover (tensor_load_to_lds) with the
//   descriptor pad feature providing the bank-conflict-avoiding LDS stride,
//   double buffered via s_wait_tensorcnt.
// - Block tile 128x128, wave tile 32x64: 8 WMMA per wave per K-step against
//   12 ds_load_b128 (1.5 LDS loads per WMMA).
// - Flash attention: 32-key chunks, full-K WMMA for both Q.K^T and P.V,
//   V pre-transposed to [b,h,d,s] for vector loads.
// ---------------------------------------------------------------------------

typedef __attribute__((ext_vector_type(16))) __bf16 v16bf;
typedef __attribute__((ext_vector_type(8)))  float  v8f;
typedef __attribute__((ext_vector_type(4)))  unsigned int v4u;
typedef __attribute__((ext_vector_type(8)))  int v8i;
typedef __attribute__((ext_vector_type(4)))  int v4i;

static constexpr int Bc  = 2;
static constexpr int Sc  = 1024;
static constexpr int Dc  = 2048;
static constexpr int Hc  = 16;
static constexpr int HDc = 128;
static constexpr int Fc  = 5632;
static constexpr int Ec  = 8;
static constexpr int Tc  = Bc * Sc;          // 2048 tokens

// ---------------------------------------------------------------------------
// Tensor Data Mover: 2D tile (tile0 x tile1 elems of 2B) from global to LDS.
// LDS padding: after every 16 DWORDs (one 32-elem row) insert 4 DWORDs
// -> effective LDS row stride = 40 bf16 (matches fragment indexing below).
// ---------------------------------------------------------------------------
static constexpr int BK   = 32;
static constexpr int STRK = BK + 8;           // padded LDS row stride (bf16)

__device__ inline void tdm_load_2d(unsigned lds_addr, const void* gaddr,
                                   unsigned dim0, unsigned dim1,
                                   unsigned tile0, unsigned tile1,
                                   unsigned stride0) {
  unsigned long long ga = (unsigned long long)(size_t)gaddr;
  v4u g0;
  g0[0] = 1u;                                           // count=1, user desc
  g0[1] = lds_addr;                                     // LDS byte address
  g0[2] = (unsigned)(ga & 0xFFFFFFFFu);                 // global addr lo
  g0[3] = (unsigned)((ga >> 32) & 0x01FFFFFFu) | 0x80000000u; // hi | type=2
  v8i g1;
  // data_size=2B (code 1), pad_enable, pad_interval=16DW (code 3),
  // pad_amount=4DW (code 3); workgroup_mask=0 (not in cluster)
  g1[0] = (int)((1u << 16) | (1u << 20) | (3u << 22) | (3u << 25));
  g1[1] = (int)((dim0 & 0xFFFFu) << 16);                // tensor_dim0 lo16
  g1[2] = (int)((dim0 >> 16) | ((dim1 & 0xFFFFu) << 16));
  g1[3] = (int)((dim1 >> 16) | (tile0 << 16));          // tile_dim0
  g1[4] = (int)(tile1);                                 // tile_dim1 (tile2=0)
  g1[5] = (int)stride0;                                 // tensor_dim0_stride
  g1[6] = 0;
  g1[7] = 0;
  v4i z4 = {0, 0, 0, 0};
#if defined(__clang_major__) && (__clang_major__ >= 23)
  v8i z8 = {0, 0, 0, 0, 0, 0, 0, 0};
  __builtin_amdgcn_tensor_load_to_lds(g0, g1, z4, z4, z8, 0);
#else
  __builtin_amdgcn_tensor_load_to_lds(g0, g1, z4, z4, 0);
#endif
}

// ---------------------------------------------------------------------------
// fp32 -> bf16 bulk conversion (grid-stride)
// ---------------------------------------------------------------------------
__global__ void cvt_f32_bf16(const float* __restrict__ in, __bf16* __restrict__ out, long n) {
  long i = (long)blockIdx.x * blockDim.x + threadIdx.x;
  long st = (long)gridDim.x * blockDim.x;
  for (; i < n; i += st) out[i] = (__bf16)in[i];
}

// ---------------------------------------------------------------------------
// Transposing fp32[K,N] -> bf16[N,K] weight conversion (tiled via LDS).
// grid (N/32, K/32), block (32,8).
// ---------------------------------------------------------------------------
__global__ __launch_bounds__(256) void transpose_cvt_kernel(const float* __restrict__ in,
                                                            __bf16* __restrict__ out,
                                                            int K, int N) {
  __shared__ float tile[32][33];
  int kt = blockIdx.y * 32, nt = blockIdx.x * 32;
  for (int i = threadIdx.y; i < 32; i += 8)
    tile[i][threadIdx.x] = in[(long)(kt + i) * N + nt + threadIdx.x];
  __syncthreads();
  for (int i = threadIdx.y; i < 32; i += 8)
    out[(long)(nt + i) * K + kt + threadIdx.x] = (__bf16)tile[threadIdx.x][i];
}

// ---------------------------------------------------------------------------
// V transpose: fp32 v[(b*S+s)*D + h*HD + d] -> bf16 vt[((b*H+h)*HD+d)*S + s]
// grid (S/32, HD/32, B*H), block (32,8).
// ---------------------------------------------------------------------------
__global__ __launch_bounds__(256) void vtrans_kernel(const float* __restrict__ vf,
                                                     __bf16* __restrict__ vt) {
  __shared__ float tile[32][33];
  int bh = blockIdx.z;
  int b = bh / Hc, h = bh % Hc;
  int st = blockIdx.x * 32, dt = blockIdx.y * 32;
  for (int i = threadIdx.y; i < 32; i += 8)
    tile[i][threadIdx.x] = vf[((long)b * Sc + st + i) * Dc + h * HDc + dt + threadIdx.x];
  __syncthreads();
  for (int i = threadIdx.y; i < 32; i += 8)
    vt[((long)bh * HDc + dt + i) * Sc + st + threadIdx.x] = (__bf16)tile[threadIdx.x][i];
}

// ---------------------------------------------------------------------------
// RMSNorm: one block per token row; writes bf16 (for WMMA) and optional fp32.
// ---------------------------------------------------------------------------
__global__ __launch_bounds__(256) void rmsnorm_kernel(const float* __restrict__ x,
                                                      const float* __restrict__ w,
                                                      __bf16* __restrict__ out_bf,
                                                      float* __restrict__ out_f) {
  int row = blockIdx.x;
  const float* xr = x + (long)row * Dc;
  __shared__ float red[256];
  float ss = 0.f;
  for (int i = threadIdx.x; i < Dc; i += 256) { float v = xr[i]; ss += v * v; }
  red[threadIdx.x] = ss;
  __syncthreads();
  for (int o = 128; o > 0; o >>= 1) {
    if ((int)threadIdx.x < o) red[threadIdx.x] += red[threadIdx.x + o];
    __syncthreads();
  }
  float rinv = rsqrtf(red[0] / (float)Dc + 1e-5f);
  for (int i = threadIdx.x; i < Dc; i += 256) {
    float v = xr[i] * rinv * w[i];
    out_bf[(long)row * Dc + i] = (__bf16)v;
    if (out_f) out_f[(long)row * Dc + i] = v;
  }
}

// ---------------------------------------------------------------------------
// Tiled bf16 WMMA GEMM with TDM-staged, double-buffered LDS tiles.
//   C[M,N](fp32) = (acc ? C : 0) + rowScale * (A[M,K] x Bt[N,K]^T)
// A row-major [M,K]; Bt row-major [N,K] (pre-transposed weights).
// Block 256 thr = 8 waves (4x2); wave tile 32x64 = 2x4 wmma; K-step 32.
// ---------------------------------------------------------------------------
static constexpr int BM = 128, BN = 128;

__global__ __launch_bounds__(256) void gemm_bf16_wmma(
    const __bf16* __restrict__ A, const __bf16* __restrict__ Bt, float* __restrict__ C,
    int M, int N, int Kd, const float* __restrict__ rowScale, int ssv, int accFlag) {
  __shared__ __align__(16) __bf16 As[2 * BM * STRK];   // 2 x 128 x 40
  __shared__ __align__(16) __bf16 Bs[2 * BN * STRK];   // 2 x 128 x 40
  int tid  = threadIdx.x;
  int lane = tid & 31, wave = tid >> 5;
  int wm = wave & 3, wn = wave >> 2;
  int hl = lane >> 4, rl = lane & 15;
  long rowBase = (long)blockIdx.y * BM;
  long colBase = (long)blockIdx.x * BN;
  int niter = Kd / BK;

  unsigned asBase = (unsigned)(size_t)(&As[0]);
  unsigned bsBase = (unsigned)(size_t)(&Bs[0]);
  const unsigned asBuf = BM * STRK * 2;                // bytes per A buffer
  const unsigned bsBuf = BN * STRK * 2;                // bytes per B buffer

  v8f acc[2][4];
  for (int i = 0; i < 2; ++i)
    for (int j = 0; j < 4; ++j)
      for (int v = 0; v < 8; ++v) acc[i][j][v] = 0.f;

  // prologue: stage tiles for k=0 into buffer 0 (wave 0 drives the TDM)
  if (tid < 32) {
    tdm_load_2d(asBase, A + rowBase * Kd, (unsigned)Kd, (unsigned)M, BK, BM, (unsigned)Kd);
    tdm_load_2d(bsBase, Bt + colBase * Kd, (unsigned)Kd, (unsigned)N, BK, BN, (unsigned)Kd);
  }

  for (int k = 0; k < niter; ++k) {
    if (tid < 32) {
      if (k + 1 < niter) {  // stage next tiles into the other buffer
        int nb = (k + 1) & 1;
        long k1 = (long)(k + 1) * BK;
        tdm_load_2d(asBase + nb * asBuf, A + rowBase * Kd + k1,
                    (unsigned)Kd, (unsigned)M, BK, BM, (unsigned)Kd);
        tdm_load_2d(bsBase + nb * bsBuf, Bt + colBase * Kd + k1,
                    (unsigned)Kd, (unsigned)N, BK, BN, (unsigned)Kd);
        __builtin_amdgcn_s_wait_tensorcnt(2);   // current pair complete
      } else {
        __builtin_amdgcn_s_wait_tensorcnt(0);   // last pair complete
      }
    }
    __syncthreads();

    const __bf16* Asb = As + (k & 1) * BM * STRK;
    const __bf16* Bsb = Bs + (k & 1) * BN * STRK;

    // A fragments: lane rl = row M; half elems j: K=j+hl*16 (contig)
    v16bf afr[2], bfr[4];
    for (int i = 0; i < 2; ++i) {
      int arow = wm * 32 + i * 16 + rl;
      for (int j = 0; j < 16; ++j) {
        int kl = j + hl * 16;
        afr[i][j] = Asb[arow * STRK + kl];
      }
    }
    // B fragments from Bt tile [n][k]: lane rl = col N; K=j+hl*16 (contiguous)
    for (int j2 = 0; j2 < 4; ++j2) {
      int brow = wn * 64 + j2 * 16 + rl;
      for (int j = 0; j < 16; ++j)
        bfr[j2][j] = Bsb[brow * STRK + hl * 16 + j];
    }
    for (int i = 0; i < 2; ++i)
      for (int j2 = 0; j2 < 4; ++j2)
        acc[i][j2] = __builtin_amdgcn_wmma_f32_16x16x32_bf16(
            false, afr[i], false, bfr[j2], (short)0, acc[i][j2], false, false);
    __syncthreads();   // all reads of this buffer done before it is re-staged
  }

  // C layout: VGPR v -> M = hl*8 + v ; N = rl
  for (int i = 0; i < 2; ++i)
    for (int j2 = 0; j2 < 4; ++j2)
      for (int v = 0; v < 8; ++v) {
        long row = rowBase + wm * 32 + i * 16 + hl * 8 + v;
        long col = colBase + wn * 64 + j2 * 16 + rl;
        float s = rowScale ? rowScale[row * ssv] : 1.0f;
        long idx = row * N + col;
        float val = acc[i][j2][v] * s + (accFlag ? C[idx] : 0.0f);
        C[idx] = val;
      }
}

// ---------------------------------------------------------------------------
// RoPE on q,k (fp32 in) + bf16 convert out. One block per (token, head).
// ---------------------------------------------------------------------------
__global__ __launch_bounds__(64) void rope_cvt_kernel(const float* __restrict__ qf,
                                                      const float* __restrict__ kf,
                                                      __bf16* __restrict__ qo,
                                                      __bf16* __restrict__ ko) {
  int th = blockIdx.x;
  int t = th / Hc, h = th % Hc;
  int s = t % Sc;
  int d = threadIdx.x;                       // 0..63
  long base = (long)t * Dc + h * HDc;
  float freq = (float)s * __powf(10000.0f, -((2.0f * d) / (float)HDc));
  float c = __cosf(freq), sn = __sinf(freq);
  float q1 = qf[base + d], q2 = qf[base + d + 64];
  float k1 = kf[base + d], k2 = kf[base + d + 64];
  qo[base + d]      = (__bf16)(q1 * c - q2 * sn);
  qo[base + d + 64] = (__bf16)(q2 * c + q1 * sn);
  ko[base + d]      = (__bf16)(k1 * c - k2 * sn);
  ko[base + d + 64] = (__bf16)(k2 * c + k1 * sn);
}

// ---------------------------------------------------------------------------
// Causal flash attention, 1 wave per 16-query tile per (b,h).
// 32-key chunks: 8 WMMA (Q.K^T, K=128) + 8 WMMA (P.V, K=32) per chunk.
// V comes pre-transposed as vt[((b*H+h)*HD + d)*S + s].
// ---------------------------------------------------------------------------
static constexpr int PSTR = 40;  // padded LDS stride for P tile (bf16)

__global__ __launch_bounds__(32) void attn_kernel(const __bf16* __restrict__ q,
                                                  const __bf16* __restrict__ k,
                                                  const __bf16* __restrict__ vt,
                                                  float* __restrict__ out) {
  int qt = blockIdx.x, h = blockIdx.y, b = blockIdx.z;
  int lane = threadIdx.x;
  int hl = lane >> 4, rl = lane & 15;
  int q0 = qt * 16;
  long qrow = ((long)b * Sc + q0) * Dc + h * HDc;
  __shared__ __align__(16) __bf16 ps[16 * PSTR];   // P tile 16x32, padded

  // Q fragments (A layout): lane rl = query row; 4 chunks over HD=128
  v16bf qfrag[4];
  for (int c = 0; c < 4; ++c)
    for (int j = 0; j < 16; ++j) {
      int kl = j + hl * 16 + c * 32;
      qfrag[c][j] = q[qrow + (long)rl * Dc + kl];
    }

  float m[8], l[8];
  v8f o[8];
  for (int i = 0; i < 8; ++i) {
    m[i] = -3.0e38f; l[i] = 0.f;
    for (int j = 0; j < 8; ++j) o[i][j] = 0.f;
  }
  const float rscale = 0.08838834764831845f;   // 1/sqrt(128)
  long vbase = ((long)b * Hc + h) * HDc;       // vt row block for this head

  for (int kb = 0; kb * 32 <= q0 + 15; ++kb) {
    // ---- scores + online softmax for two 16-key sub-tiles ----
    for (int kg = 0; kg < 2; ++kg) {
      int key16 = kb * 32 + kg * 16;
      if (key16 <= q0 + 15) {
        long kbase = ((long)b * Sc + key16) * Dc + h * HDc;
        v8f sacc;
        for (int i = 0; i < 8; ++i) sacc[i] = 0.f;
        for (int c = 0; c < 4; ++c) {
          v16bf kfrag = *(const v16bf*)&k[kbase + (long)rl * Dc + c * 32 + hl * 16];
          sacc = __builtin_amdgcn_wmma_f32_16x16x32_bf16(
              false, qfrag[c], false, kfrag, (short)0, sacc, false, false);
        }
        for (int vv = 0; vv < 8; ++vv) {
          int col = key16 + rl;
          int row = q0 + hl * 8 + vv;
          bool masked = col > row;
          float sv = masked ? -3.0e38f : sacc[vv] * rscale;
          float mx = sv;
          for (int off = 1; off <= 8; off <<= 1) mx = fmaxf(mx, __shfl_xor(mx, off, 32));
          float mnew = fmaxf(m[vv], mx);
          float p = masked ? 0.f : __expf(sv - mnew);
          float sum = p;
          for (int off = 1; off <= 8; off <<= 1) sum += __shfl_xor(sum, off, 32);
          float alpha = __expf(m[vv] - mnew);
          m[vv] = mnew;
          l[vv] = l[vv] * alpha + sum;
          for (int n = 0; n < 8; ++n) o[n][vv] *= alpha;
          ps[(hl * 8 + vv) * PSTR + kg * 16 + rl] = (__bf16)p;
        }
      } else {
        for (int vv = 0; vv < 8; ++vv)
          ps[(hl * 8 + vv) * PSTR + kg * 16 + rl] = (__bf16)0.0f;
      }
    }
    __syncthreads();
    // ---- P (16x32) as full-K A fragment; V fragments are contiguous ----
    v16bf pa;
    for (int j = 0; j < 16; ++j) {
      int kk = j + hl * 16;
      pa[j] = ps[rl * PSTR + kk];
    }
    for (int n = 0; n < 8; ++n) {
      // B fragment of V chunk (32 x 16): lane rl = dim col; K contiguous
      v16bf vb = *(const v16bf*)&vt[(vbase + n * 16 + rl) * Sc + kb * 32 + hl * 16];
      o[n] = __builtin_amdgcn_wmma_f32_16x16x32_bf16(
          false, pa, false, vb, (short)0, o[n], false, false);
    }
    __syncthreads();
  }

  for (int n = 0; n < 8; ++n)
    for (int vv = 0; vv < 8; ++vv) {
      long row = (long)b * Sc + q0 + hl * 8 + vv;
      out[row * Dc + h * HDc + n * 16 + rl] = o[n][vv] / l[vv];
    }
}

// ---------------------------------------------------------------------------
// Router: 1 block (8 waves) per token; wave e computes logit_e; lane reduce;
// then softmax + top-2 + normalized dense combine weights (wfull[T,E]).
// ---------------------------------------------------------------------------
__global__ __launch_bounds__(256) void router_kernel(const float* __restrict__ x,
                                                     const float* __restrict__ wr,
                                                     float* __restrict__ logits_out,
                                                     float* __restrict__ wfull) {
  int t = blockIdx.x;
  int lane = threadIdx.x & 31, wave = threadIdx.x >> 5;  // wave == expert
  __shared__ float slog[Ec];
  const float* xr = x + (long)t * Dc;
  const float* we = wr + (long)wave * Dc;
  float s = 0.f;
  for (int i = lane; i < Dc; i += 32) s += xr[i] * we[i];
  for (int off = 16; off > 0; off >>= 1) s += __shfl_xor(s, off, 32);
  if (lane == 0) slog[wave] = s;
  __syncthreads();
  if (threadIdx.x == 0) {
    float mx = slog[0];
    for (int e = 1; e < Ec; ++e) mx = fmaxf(mx, slog[e]);
    float p[Ec], sum = 0.f;
    for (int e = 0; e < Ec; ++e) { p[e] = __expf(slog[e] - mx); sum += p[e]; }
    for (int e = 0; e < Ec; ++e) p[e] /= sum;
    int i1 = 0;
    for (int e = 1; e < Ec; ++e) if (p[e] > p[i1]) i1 = e;
    int i2 = (i1 == 0) ? 1 : 0;
    for (int e = 0; e < Ec; ++e) if (e != i1 && p[e] > p[i2]) i2 = e;
    float wsum = p[i1] + p[i2];
    for (int e = 0; e < Ec; ++e) wfull[(long)t * Ec + e] = 0.f;
    wfull[(long)t * Ec + i1] = p[i1] / wsum;
    wfull[(long)t * Ec + i2] = p[i2] / wsum;
    for (int e = 0; e < Ec; ++e) logits_out[(long)t * Ec + e] = slog[e];
  }
}

// ---------------------------------------------------------------------------
// h = bf16( silu(g) * u )
// ---------------------------------------------------------------------------
__global__ void silu_mul_kernel(const float* __restrict__ g, const float* __restrict__ u,
                                __bf16* __restrict__ h, long n) {
  long i = (long)blockIdx.x * blockDim.x + threadIdx.x;
  long st = (long)gridDim.x * blockDim.x;
  for (; i < n; i += st) {
    float gv = g[i];
    h[i] = (__bf16)((gv / (1.f + __expf(-gv))) * u[i]);
  }
}

// ---------------------------------------------------------------------------
extern "C" void kernel_launch(void* const* d_in, const int* in_sizes, int n_in,
                              void* d_out, int out_size, void* d_ws, size_t ws_size,
                              hipStream_t stream) {
  (void)in_sizes; (void)n_in; (void)out_size; (void)ws_size;
  const float* hs      = (const float*)d_in[0];
  const float* wq      = (const float*)d_in[1];
  const float* wk      = (const float*)d_in[2];
  const float* wv      = (const float*)d_in[3];
  const float* wo      = (const float*)d_in[4];
  const float* w_inln  = (const float*)d_in[5];
  const float* w_pln   = (const float*)d_in[6];
  const float* w_rout  = (const float*)d_in[7];
  const float* wg      = (const float*)d_in[8];
  const float* wu      = (const float*)d_in[9];
  const float* wd      = (const float*)d_in[10];

  float* out_hidden = (float*)d_out;                    // [T, D]
  float* out_logits = out_hidden + (long)Tc * Dc;       // [T, E]

  // ---- workspace carve ----
  char* p = (char*)d_ws;
  auto alloc = [&](size_t bytes) -> void* {
    void* r = (void*)p;
    p += (bytes + 255) & ~(size_t)255;
    return r;
  };
  __bf16* wqT    = (__bf16*)alloc((size_t)Dc * Dc * 2);   // [N,K] transposed
  __bf16* wkT    = (__bf16*)alloc((size_t)Dc * Dc * 2);
  __bf16* wvT    = (__bf16*)alloc((size_t)Dc * Dc * 2);
  __bf16* woT    = (__bf16*)alloc((size_t)Dc * Dc * 2);
  __bf16* wgT_e  = (__bf16*)alloc((size_t)Dc * Fc * 2);   // per-expert, reused
  __bf16* wuT_e  = (__bf16*)alloc((size_t)Dc * Fc * 2);
  __bf16* wdT_e  = (__bf16*)alloc((size_t)Fc * Dc * 2);
  __bf16* xn_bf  = (__bf16*)alloc((size_t)Tc * Dc * 2);
  float*  q_f    = (float*)alloc((size_t)Tc * Dc * 4);
  float*  k_f    = (float*)alloc((size_t)Tc * Dc * 4);
  float*  v_f    = (float*)alloc((size_t)Tc * Dc * 4);
  __bf16* qr_bf  = (__bf16*)alloc((size_t)Tc * Dc * 2);
  __bf16* kr_bf  = (__bf16*)alloc((size_t)Tc * Dc * 2);
  __bf16* vt_bf  = (__bf16*)alloc((size_t)Tc * Dc * 2);   // [b,h,d,s]
  float*  attn_f = (float*)alloc((size_t)Tc * Dc * 4);
  __bf16* attn_b = (__bf16*)alloc((size_t)Tc * Dc * 2);
  __bf16* x2_bf  = (__bf16*)alloc((size_t)Tc * Dc * 2);
  float*  x2_f   = (float*)alloc((size_t)Tc * Dc * 4);
  float*  wfull  = (float*)alloc((size_t)Tc * Ec * 4);
  float*  g_f    = (float*)alloc((size_t)Tc * Fc * 4);
  float*  u_f    = (float*)alloc((size_t)Tc * Fc * 4);
  __bf16* h_bf   = (__bf16*)alloc((size_t)Tc * Fc * 2);

  dim3 cvtG(2048), cvtB(256);
  dim3 tb(32, 8);
  auto gemm = [&](const __bf16* A, const __bf16* Bt, float* C, int M, int N, int Kd,
                  const float* scale, int ssv, int acc) {
    dim3 grid(N / BN, M / BM);
    gemm_bf16_wmma<<<grid, dim3(256), 0, stream>>>(A, Bt, C, M, N, Kd, scale, ssv, acc);
  };

  // ---- attention block ----
  transpose_cvt_kernel<<<dim3(Dc / 32, Dc / 32), tb, 0, stream>>>(wq, wqT, Dc, Dc);
  transpose_cvt_kernel<<<dim3(Dc / 32, Dc / 32), tb, 0, stream>>>(wk, wkT, Dc, Dc);
  transpose_cvt_kernel<<<dim3(Dc / 32, Dc / 32), tb, 0, stream>>>(wv, wvT, Dc, Dc);
  transpose_cvt_kernel<<<dim3(Dc / 32, Dc / 32), tb, 0, stream>>>(wo, woT, Dc, Dc);

  rmsnorm_kernel<<<Tc, 256, 0, stream>>>(hs, w_inln, xn_bf, (float*)nullptr);

  gemm(xn_bf, wqT, q_f, Tc, Dc, Dc, nullptr, 0, 0);
  gemm(xn_bf, wkT, k_f, Tc, Dc, Dc, nullptr, 0, 0);
  gemm(xn_bf, wvT, v_f, Tc, Dc, Dc, nullptr, 0, 0);

  rope_cvt_kernel<<<Tc * Hc, 64, 0, stream>>>(q_f, k_f, qr_bf, kr_bf);
  vtrans_kernel<<<dim3(Sc / 32, HDc / 32, Bc * Hc), tb, 0, stream>>>(v_f, vt_bf);

  attn_kernel<<<dim3(Sc / 16, Hc, Bc), 32, 0, stream>>>(qr_bf, kr_bf, vt_bf, attn_f);
  cvt_f32_bf16<<<cvtG, cvtB, 0, stream>>>(attn_f, attn_b, (long)Tc * Dc);

  // residual + O projection (accumulate into residual copy)
  hipMemcpyAsync(out_hidden, hs, (size_t)Tc * Dc * 4, hipMemcpyDeviceToDevice, stream);
  gemm(attn_b, woT, out_hidden, Tc, Dc, Dc, nullptr, 0, 1);

  // ---- MoE block ----
  rmsnorm_kernel<<<Tc, 256, 0, stream>>>(out_hidden, w_pln, x2_bf, x2_f);
  router_kernel<<<Tc, 256, 0, stream>>>(x2_f, w_rout, out_logits, wfull);

  for (int e = 0; e < Ec; ++e) {
    transpose_cvt_kernel<<<dim3(Fc / 32, Dc / 32), tb, 0, stream>>>(
        wg + (long)e * Dc * Fc, wgT_e, Dc, Fc);
    transpose_cvt_kernel<<<dim3(Fc / 32, Dc / 32), tb, 0, stream>>>(
        wu + (long)e * Dc * Fc, wuT_e, Dc, Fc);
    transpose_cvt_kernel<<<dim3(Dc / 32, Fc / 32), tb, 0, stream>>>(
        wd + (long)e * Fc * Dc, wdT_e, Fc, Dc);
    gemm(x2_bf, wgT_e, g_f, Tc, Fc, Dc, nullptr, 0, 0);
    gemm(x2_bf, wuT_e, u_f, Tc, Fc, Dc, nullptr, 0, 0);
    silu_mul_kernel<<<cvtG, cvtB, 0, stream>>>(g_f, u_f, h_bf, (long)Tc * Fc);
    // out += (h @ wd[e]) * wfull[:, e]   (accumulate into residual)
    gemm(h_bf, wdT_e, out_hidden, Tc, Dc, Fc, wfull + e, Ec, 1);
  }
}
